// InterfaceAttentionBlock_58548994179273
// MI455X (gfx1250) — matrix operand order repaired and checked by reading, not compiled, not run on hardware
//
#include <hip/hip_runtime.h>
#include <hip/hip_bf16.h>

// ---------------------------------------------------------------------------
// MI455X (gfx1250) fused InterfaceAttentionBlock.
// Roofline: ~0.95 TFLOP vs ~530 MB HBM traffic -> compute bound unless the
// per-row MLP runs on WMMA. We tile 32 rows per workgroup (two 16-row WMMA
// M tiles) so each packed B-fragment load from L2 feeds two v_wmma ops and
// the per-block 0.95 MB weight stream is amortized over 32 rows (~30 GB of
// L2 traffic total, vs ~0.2-0.4 ms of WMMA math).
// ---------------------------------------------------------------------------

typedef __attribute__((ext_vector_type(16))) _Float16 v16h;
typedef __attribute__((ext_vector_type(8)))  _Float16 v8h;
typedef __attribute__((ext_vector_type(8)))  float    v8f;

#define TM   32          // rows per workgroup (two WMMA M tiles)
#define NT   256         // 8 wave32 per workgroup
#define HD   256         // hidden dim
#define DIN  192         // TB1 input dim

#define KT1  6           // 192/32 k-tiles
#define KT2  8           // 256/32 k-tiles
#define FRAG_H 512       // halves per fragment: 32 lanes * 16 halves

// packed-weight sub-offsets (in halves)
#define MAT_H_192 (16*KT1*FRAG_H)   // 49152
#define MAT_H_256 (16*KT2*FRAG_H)   // 65536
#define H_SW1  0
#define H_SS1  (H_SW1  + MAT_H_192)
#define H_SIG1 (H_SS1  + MAT_H_192)
#define H_SC1  (H_SIG1 + MAT_H_192)
#define H_SW2  (H_SC1  + MAT_H_256)
#define H_SS2  (H_SW2  + MAT_H_256)
#define H_SIG2 (H_SS2  + MAT_H_256)
#define H_SC2  (H_SIG2 + MAT_H_256)
#define H_TOTAL (H_SC2 + MAT_H_256) // 475136 halves

// -------------------------------- helpers ----------------------------------

__device__ __forceinline__ float sigm(float x) { return 1.0f / (1.0f + __expf(-x)); }

// order-preserving float->uint map for atomicMax-based segment max
__device__ __forceinline__ unsigned int ordf(float f) {
    unsigned int u = __float_as_uint(f);
    return (u & 0x80000000u) ? ~u : (u | 0x80000000u);
}
__device__ __forceinline__ float unordf(unsigned int u) {
    return __uint_as_float((u & 0x80000000u) ? (u ^ 0x80000000u) : ~u);
}

__device__ __forceinline__ v8f wmma16(v16h a, v16h b, v8f c) {
    // D = A(16x32 f16) x B(32x16 f16) + C(16x16 f32)
    return __builtin_amdgcn_wmma_f32_16x16x32_f16(
        /*neg_a=*/false, a, /*neg_b=*/false, b,
        /*c_mod=*/(short)0, c, /*reuse_a=*/false, /*reuse_b=*/false);
}

// Build A fragment from LDS row-major [TM][strideH] halves.
// Documented 16-bit A layout: lane L (m = L&15, hi = L>>4):
//   halves 0..7  -> K = kbase + hi*8 + 0..7
//   halves 8..15 -> K = kbase + 16 + hi*8 + 0..7
// -> two 16B contiguous LDS loads per fragment (ds_load_b128).
__device__ __forceinline__ v16h lds_a_frag(const _Float16* base, int strideH,
                                           int m, int hi, int kbase) {
    const _Float16* p = base + (size_t)m * strideH + kbase + hi * 8;
    v8h lo = *(const v8h*)p;
    v8h up = *(const v8h*)(p + 16);
    v16h r;
#pragma unroll
    for (int i = 0; i < 8; ++i) { r[i] = lo[i]; r[i + 8] = up[i]; }
    return r;
}

// ----------------------------- prep kernels --------------------------------

// Repack W[out=256][K] f32 (row-major) into f16 B-fragment-linear layout:
// dst[((ct*Kt + kt)*32 + lane)*16 + i] = W[n][k],
//   n = ct*16 + (lane&15),  k = kt*32 + (i&7) + (i>=8 ? 16 : 0) + (lane>=16 ? 8 : 0)
// (B 32x16 layout matches A: lanes 0-15 hold K 0..7,16..23; lanes 16-31 hold K 8..15,24..31.)
__global__ void pack_weight_kernel(const float* __restrict__ src,
                                   _Float16* __restrict__ dst,
                                   int K, int total) {
    int idx = blockIdx.x * blockDim.x + threadIdx.x;
    if (idx >= total) return;
    int Kt     = K >> 5;
    int frag   = idx >> 9;
    int within = idx & 511;
    int lane   = within >> 4;
    int i      = within & 15;
    int ct = frag / Kt;
    int kt = frag - ct * Kt;
    int n = ct * 16 + (lane & 15);
    int k = kt * 32 + (i & 7) + ((i >> 3) << 4) + ((lane >= 16) ? 8 : 0);
    dst[idx] = (_Float16)src[(size_t)n * K + k];
}

__global__ void zero_f32_kernel(float* __restrict__ p, long long n) {
    long long i = (long long)blockIdx.x * blockDim.x + threadIdx.x;
    if (i < n) p[i] = 0.0f;
}

// --------------------------- fused main kernel -----------------------------

__global__ __launch_bounds__(NT)
void fused_forward_kernel(const float* __restrict__ veq,
                          const float* __restrict__ vaf,
                          const int*   __restrict__ gidx,
                          const float* __restrict__ fw,
                          const float* __restrict__ fb,
                          const float* __restrict__ ln1g,
                          const float* __restrict__ ln1b,
                          const float* __restrict__ bsig1,
                          const float* __restrict__ ln2g,
                          const float* __restrict__ ln2b,
                          const float* __restrict__ bsig2,
                          const float* __restrict__ outw,
                          const float* __restrict__ outb,
                          const _Float16* __restrict__ pw,
                          float* __restrict__ scores,
                          unsigned int* __restrict__ smax,
                          long long N) {
    __shared__ __align__(16) _Float16 sh_xln[TM][264]; // LN output (f16), TB1 then TB2
    __shared__ __align__(16) _Float16 sh_b[TM][264];   // gated branch b (f16)
    __shared__ __align__(16) float    sh_x[TM][264];   // raw x -> TB1 out -> TB2 out
    __shared__ float sh_ow[HD * 4];
    __shared__ float sh_ob[4];
    __shared__ float sh_red[TM][8][2];
    __shared__ float sh_stat[TM][2];

    const int tid = threadIdx.x;
    const long long row0 = (long long)blockIdx.x * TM;

    for (int i = tid; i < HD * 4; i += NT) sh_ow[i] = outw[i];
    if (tid < 4) sh_ob[tid] = outb[tid];

    // ---- Stage 0: fourier + concat + LayerNorm(192) -> sh_xln (f16) ----
    {
        int r = tid >> 3, c = tid & 7;           // 8 threads per row, 24 feats each
        long long g = row0 + r;
        bool valid = g < N;
        float vx = 0.f, vy = 0.f, vz = 0.f;
        if (valid) { vx = veq[g*3+0]; vy = veq[g*3+1]; vz = veq[g*3+2]; }
        float nrm = sqrtf(vx*vx + vy*vy + vz*vz);
        float s = 0.f, s2 = 0.f;
#pragma unroll
        for (int j = 0; j < 24; ++j) {
            int f = c * 24 + j;
            float x;
            if (f < 64) x = valid ? cosf(6.28318530718f * (nrm * fw[f] + fb[f])) : 0.f;
            else        x = valid ? vaf[g * 128 + (f - 64)] : 0.f;
            sh_x[r][f] = x;
            s += x; s2 += x * x;
        }
        sh_red[r][c][0] = s; sh_red[r][c][1] = s2;
    }
    __syncthreads();
    {
        int r = tid >> 3, c = tid & 7;
        if (c == 0) {
            float s = 0.f, s2 = 0.f;
            for (int k = 0; k < 8; ++k) { s += sh_red[r][k][0]; s2 += sh_red[r][k][1]; }
            float mu = s / 192.f;
            float var = s2 / 192.f - mu * mu;
            sh_stat[r][0] = mu; sh_stat[r][1] = rsqrtf(var + 1e-5f);
        }
    }
    __syncthreads();
    {
        int r = tid >> 3, c = tid & 7;
        float mu = sh_stat[r][0], rs = sh_stat[r][1];
#pragma unroll
        for (int j = 0; j < 24; ++j) {
            int f = c * 24 + j;
            sh_xln[r][f] = (_Float16)((sh_x[r][f] - mu) * rs * ln1g[f] + ln1b[f]);
        }
    }
    __syncthreads();

    const int wv = tid >> 5, lane = tid & 31;
    const int m16 = lane & 15, hi = lane >> 4;

    // ---- TB1: sw/ss/sig GEMMs (K=192), two 16-row M tiles ----
    float sig1[2][2][8];                            // [cti][rt][r]
    for (int cti = 0; cti < 2; ++cti) {
        int ct = wv + cti * 8;                      // column tile 0..15
        v8f a_sw[2] = {}, a_ss[2] = {}, a_sg[2] = {};
        for (int kt = 0; kt < KT1; ++kt) {
            v16h a0 = lds_a_frag(&sh_xln[0][0], 264, m16,      hi, kt * 32);
            v16h a1 = lds_a_frag(&sh_xln[0][0], 264, 16 + m16, hi, kt * 32);
            int fo = ((ct * KT1 + kt) * 32 + lane) * 16;
            v16h bsw = *(const v16h*)(pw + H_SW1  + fo);
            v16h bss = *(const v16h*)(pw + H_SS1  + fo);
            v16h bsg = *(const v16h*)(pw + H_SIG1 + fo);
            a_sw[0] = wmma16(a0, bsw, a_sw[0]);  a_sw[1] = wmma16(a1, bsw, a_sw[1]);
            a_ss[0] = wmma16(a0, bss, a_ss[0]);  a_ss[1] = wmma16(a1, bss, a_ss[1]);
            a_sg[0] = wmma16(a0, bsg, a_sg[0]);  a_sg[1] = wmma16(a1, bsg, a_sg[1]);
        }
        int ncol = ct * 16 + m16;
        float bs = bsig1[ncol];
#pragma unroll
        for (int rt = 0; rt < 2; ++rt) {
#pragma unroll
            for (int r = 0; r < 8; ++r) {           // D layout: m = hi*8 + r
                int m = rt * 16 + hi * 8 + r;
                float swv = a_sw[rt][r];
                float bv = swv * sigm(swv) * a_ss[rt][r]; // silu(x@Wsw)*(x@Wss)
                sh_b[m][ncol] = (_Float16)bv;
                sig1[cti][rt][r] = a_sg[rt][r] + bs;
            }
        }
    }
    __syncthreads();

    // ---- TB1: y = sigmoid(sig) * (b @ Wscale^T) -> sh_x (f32) ----
    for (int cti = 0; cti < 2; ++cti) {
        int ct = wv + cti * 8;
        v8f acc[2] = {};
        for (int kt = 0; kt < KT2; ++kt) {
            v16h a0 = lds_a_frag(&sh_b[0][0], 264, m16,      hi, kt * 32);
            v16h a1 = lds_a_frag(&sh_b[0][0], 264, 16 + m16, hi, kt * 32);
            int fo = ((ct * KT2 + kt) * 32 + lane) * 16;
            v16h b = *(const v16h*)(pw + H_SC1 + fo);
            acc[0] = wmma16(a0, b, acc[0]);  acc[1] = wmma16(a1, b, acc[1]);
        }
        int ncol = ct * 16 + m16;
#pragma unroll
        for (int rt = 0; rt < 2; ++rt)
#pragma unroll
            for (int r = 0; r < 8; ++r) {
                int m = rt * 16 + hi * 8 + r;
                sh_x[m][ncol] = sigm(sig1[cti][rt][r]) * acc[rt][r]; // no residual
            }
    }
    __syncthreads();

    // ---- LayerNorm(256) -> sh_xln (f16) ----
    {
        int r = tid >> 3, c = tid & 7;
        float s = 0.f, s2 = 0.f;
#pragma unroll
        for (int j = 0; j < 32; ++j) { float x = sh_x[r][c*32 + j]; s += x; s2 += x*x; }
        sh_red[r][c][0] = s; sh_red[r][c][1] = s2;
    }
    __syncthreads();
    {
        int r = tid >> 3, c = tid & 7;
        if (c == 0) {
            float s = 0.f, s2 = 0.f;
            for (int k = 0; k < 8; ++k) { s += sh_red[r][k][0]; s2 += sh_red[r][k][1]; }
            float mu = s / 256.f, var = s2 / 256.f - mu * mu;
            sh_stat[r][0] = mu; sh_stat[r][1] = rsqrtf(var + 1e-5f);
        }
    }
    __syncthreads();
    {
        int r = tid >> 3, c = tid & 7;
        float mu = sh_stat[r][0], rs = sh_stat[r][1];
#pragma unroll
        for (int j = 0; j < 32; ++j) {
            int f = c * 32 + j;
            sh_xln[r][f] = (_Float16)((sh_x[r][f] - mu) * rs * ln2g[f] + ln2b[f]);
        }
    }
    __syncthreads();

    // ---- TB2: sw/ss/sig GEMMs (K=256) ----
    float sig2[2][2][8];
    for (int cti = 0; cti < 2; ++cti) {
        int ct = wv + cti * 8;
        v8f a_sw[2] = {}, a_ss[2] = {}, a_sg[2] = {};
        for (int kt = 0; kt < KT2; ++kt) {
            v16h a0 = lds_a_frag(&sh_xln[0][0], 264, m16,      hi, kt * 32);
            v16h a1 = lds_a_frag(&sh_xln[0][0], 264, 16 + m16, hi, kt * 32);
            int fo = ((ct * KT2 + kt) * 32 + lane) * 16;
            v16h bsw = *(const v16h*)(pw + H_SW2  + fo);
            v16h bss = *(const v16h*)(pw + H_SS2  + fo);
            v16h bsg = *(const v16h*)(pw + H_SIG2 + fo);
            a_sw[0] = wmma16(a0, bsw, a_sw[0]);  a_sw[1] = wmma16(a1, bsw, a_sw[1]);
            a_ss[0] = wmma16(a0, bss, a_ss[0]);  a_ss[1] = wmma16(a1, bss, a_ss[1]);
            a_sg[0] = wmma16(a0, bsg, a_sg[0]);  a_sg[1] = wmma16(a1, bsg, a_sg[1]);
        }
        int ncol = ct * 16 + m16;
        float bs = bsig2[ncol];
#pragma unroll
        for (int rt = 0; rt < 2; ++rt)
#pragma unroll
            for (int r = 0; r < 8; ++r) {
                int m = rt * 16 + hi * 8 + r;
                float swv = a_sw[rt][r];
                float bv = swv * sigm(swv) * a_ss[rt][r];
                sh_b[m][ncol] = (_Float16)bv;
                sig2[cti][rt][r] = a_sg[rt][r] + bs;
            }
    }
    __syncthreads();

    // ---- TB2: x = x + sigmoid(sig) * (b @ Wscale^T) (in-place, per-lane) ----
    for (int cti = 0; cti < 2; ++cti) {
        int ct = wv + cti * 8;
        v8f acc[2] = {};
        for (int kt = 0; kt < KT2; ++kt) {
            v16h a0 = lds_a_frag(&sh_b[0][0], 264, m16,      hi, kt * 32);
            v16h a1 = lds_a_frag(&sh_b[0][0], 264, 16 + m16, hi, kt * 32);
            int fo = ((ct * KT2 + kt) * 32 + lane) * 16;
            v16h b = *(const v16h*)(pw + H_SC2 + fo);
            acc[0] = wmma16(a0, b, acc[0]);  acc[1] = wmma16(a1, b, acc[1]);
        }
        int ncol = ct * 16 + m16;
#pragma unroll
        for (int rt = 0; rt < 2; ++rt)
#pragma unroll
            for (int r = 0; r < 8; ++r) {
                int m = rt * 16 + hi * 8 + r;
                sh_x[m][ncol] = sh_x[m][ncol] + sigm(sig2[cti][rt][r]) * acc[rt][r];
            }
    }
    __syncthreads();

    // ---- out linear 256 -> 4, write scores + segment max ----
    if (tid < TM * 4) {
        int r = tid >> 2, h = tid & 3;
        long long g = row0 + r;
        if (g < N) {
            float s = sh_ob[h];
            const float* wr = &sh_ow[h * HD];
#pragma unroll 8
            for (int k = 0; k < HD; ++k) s += sh_x[r][k] * wr[k];
            scores[g * 4 + h] = s;
            int seg = gidx[g];
            atomicMax(&smax[(size_t)seg * 4 + h], ordf(s));
        }
    }
}

// --------------------------- softmax / pooling -----------------------------

__global__ void seg_exp_sum_kernel(const int* __restrict__ gidx,
                                   const unsigned int* __restrict__ smax,
                                   float* __restrict__ scores,
                                   float* __restrict__ ssum, long long N4) {
    long long i = (long long)blockIdx.x * blockDim.x + threadIdx.x;
    if (i >= N4) return;
    long long row = i >> 2;
    int h = (int)(i & 3);
    int g = gidx[row];
    float m = unordf(smax[(size_t)g * 4 + h]);
    float e = __expf(scores[i] - m);
    scores[i] = e;
    atomicAdd(&ssum[(size_t)g * 4 + h], e);
}

__global__ void seg_pool_kernel(const float* __restrict__ veq,
                                const int* __restrict__ gidx,
                                const float* __restrict__ scores,
                                const float* __restrict__ ssum,
                                float* __restrict__ out, long long N) {
    long long row = (long long)blockIdx.x * blockDim.x + threadIdx.x;
    if (row >= N) return;
    int g = gidx[row];
    float vx = veq[row*3+0], vy = veq[row*3+1], vz = veq[row*3+2];
#pragma unroll
    for (int h = 0; h < 4; ++h) {
        float a = scores[row * 4 + h] / ssum[(size_t)g * 4 + h];
        atomicAdd(&out[(size_t)g * 12 + h * 3 + 0], vx * a);
        atomicAdd(&out[(size_t)g * 12 + h * 3 + 1], vy * a);
        atomicAdd(&out[(size_t)g * 12 + h * 3 + 2], vz * a);
    }
}

// ------------------------------- launcher ----------------------------------

extern "C" void kernel_launch(void* const* d_in, const int* in_sizes, int n_in,
                              void* d_out, int out_size, void* d_ws, size_t ws_size,
                              hipStream_t stream) {
    (void)n_in; (void)ws_size;
    const float* veq    = (const float*)d_in[0];
    const float* vaf    = (const float*)d_in[1];
    const int*   gidx   = (const int*)  d_in[2];
    // d_in[3] = num_edges (device scalar; E derived from out_size instead)
    const float* fw     = (const float*)d_in[4];
    const float* fb     = (const float*)d_in[5];
    const float* ln1g   = (const float*)d_in[6];
    const float* ln1b   = (const float*)d_in[7];
    const float* w_sw1  = (const float*)d_in[8];
    const float* w_ss1  = (const float*)d_in[9];
    const float* w_sig1 = (const float*)d_in[10];
    const float* b_sig1 = (const float*)d_in[11];
    const float* w_sc1  = (const float*)d_in[12];
    const float* ln2g   = (const float*)d_in[13];
    const float* ln2b   = (const float*)d_in[14];
    const float* w_sw2  = (const float*)d_in[15];
    const float* w_ss2  = (const float*)d_in[16];
    const float* w_sig2 = (const float*)d_in[17];
    const float* b_sig2 = (const float*)d_in[18];
    const float* w_sc2  = (const float*)d_in[19];
    const float* outw   = (const float*)d_in[20];
    const float* outb   = (const float*)d_in[21];

    const long long N = (long long)in_sizes[0] / 3;
    const long long E = (long long)out_size / 12;

    char* ws = (char*)d_ws;
    size_t off = 0;
    float*        scores = (float*)(ws + off);
    off += ((size_t)N * 4 * sizeof(float) + 255) & ~(size_t)255;
    unsigned int* smax   = (unsigned int*)(ws + off);
    off += ((size_t)E * 4 * sizeof(unsigned int) + 255) & ~(size_t)255;
    float*        ssum   = (float*)(ws + off);
    off += ((size_t)E * 4 * sizeof(float) + 255) & ~(size_t)255;
    _Float16*     pw     = (_Float16*)(ws + off);   // H_TOTAL halves

    // 1) pack all eight weight matrices into WMMA fragment layout (f16)
    auto pack = [&](const float* src, size_t dstOffH, int K) {
        int total = 16 * (K >> 5) * FRAG_H;
        pack_weight_kernel<<<(total + 255) / 256, 256, 0, stream>>>(
            src, pw + dstOffH, K, total);
    };
    pack(w_sw1,  H_SW1,  DIN); pack(w_ss1, H_SS1, DIN); pack(w_sig1, H_SIG1, DIN);
    pack(w_sc1,  H_SC1,  HD);
    pack(w_sw2,  H_SW2,  HD);  pack(w_ss2, H_SS2, HD);  pack(w_sig2, H_SIG2, HD);
    pack(w_sc2,  H_SC2,  HD);

    // 2) zero segment max (ord(f) > 0 for all reals, so 0 == -inf), sums, out
    zero_f32_kernel<<<(int)((E * 4  + 255) / 256), 256, 0, stream>>>((float*)smax, E * 4);
    zero_f32_kernel<<<(int)((E * 4  + 255) / 256), 256, 0, stream>>>(ssum,         E * 4);
    zero_f32_kernel<<<(int)((E * 12 + 255) / 256), 256, 0, stream>>>((float*)d_out, E * 12);

    // 3) fused feature/MLP/score kernel (WMMA)
    long long nblk = (N + TM - 1) / TM;
    fused_forward_kernel<<<(int)nblk, NT, 0, stream>>>(
        veq, vaf, gidx, fw, fb, ln1g, ln1b, b_sig1, ln2g, ln2b, b_sig2,
        outw, outb, pw, scores, smax, N);

    // 4) segmented softmax: e = exp(s - max), segment sums
    seg_exp_sum_kernel<<<(int)((N * 4 + 255) / 256), 256, 0, stream>>>(
        gidx, smax, scores, ssum, N * 4);

    // 5) weighted segmented pooling of 3-vectors -> [E, 12]
    seg_pool_kernel<<<(int)((N + 255) / 256), 256, 0, stream>>>(
        veq, gidx, scores, ssum, (float*)d_out, N);
}
